// Transcriber_29420525978183
// MI455X (gfx1250) — compile-verified, matrix operand order checked by reading, not run on hardware
//
#include <hip/hip_runtime.h>
#include <hip/hip_bf16.h>
#include <math.h>

// ---------------------------------------------------------------------------
// B=4 T=1024 F_IN=229 ; D_MODEL=512 D_STATE=16 D_CONV=4 N_BLOCKS=4 OUT=88
// D_INNER=1024 DT_RANK=32 ; C1=32 C3=64 ; FC_IN=3648 ; M=4096 rows everywhere.
// All GEMMs run on v_wmma_f32_16x16x32_f16 with f16-staged operands.
// ---------------------------------------------------------------------------

typedef __attribute__((ext_vector_type(16))) _Float16 v16h;
typedef __attribute__((ext_vector_type(8)))  _Float16 v8h;
typedef __attribute__((ext_vector_type(4)))  _Float16 v4h;
typedef __attribute__((ext_vector_type(8)))  float    v8f;

constexpr int EPI_NONE          = 0;
constexpr int EPI_BIAS          = 1;
constexpr int EPI_BIAS_SIGMOID  = 2;
constexpr int EPI_BIAS_SOFTPLUS = 3;

// OMODE bits: 1 = write f32 C, 2 = write f16 C
constexpr int OUT_F32  = 1;
constexpr int OUT_F16  = 2;
constexpr int OUT_BOTH = 3;

// Lane fragment (documented CDNA5 16-bit layout): half h owns K =
// {kb..kb+7, kb+16..kb+23}, kb = 8h.  With f16 operands that is two b128
// loads, concatenated at register level (no VALU).
__device__ __forceinline__ v16h load_frag_h(const _Float16* __restrict__ p) {
    v8h lo = *(const v8h*)(p);
    v8h hi = *(const v8h*)(p + 16);
    return __builtin_shufflevector(lo, hi, 0,1,2,3,4,5,6,7,8,9,10,11,12,13,14,15);
}

// ---------------------------------------------------------------------------
// WMMA GEMM:  C[M,N] = A[M,K] @ W[N,K]^T  (f16 in, f32 accumulate)
// One wave computes a 16x(16*NT) strip.  The K-loop is unrolled 2x into a
// register ping-pong: buffer set 0 feeds WMMA while set 1 loads, then the
// roles flip — no buffer-rotation copies, loads clause up, waits stagger.
// Requires K % 32 == 0, M % 16 == 0 (true at every call site).
// ---------------------------------------------------------------------------
template<int EPI, int OMODE, int NT>
__global__ __launch_bounds__(256)
void gemm_wmma_f16(const _Float16* __restrict__ A, int lda,
                   const _Float16* __restrict__ W, int ldw,
                   const float* __restrict__ bias,
                   float* __restrict__ C32, _Float16* __restrict__ C16, int ldc,
                   int M, int N, int K)
{
    const int lane   = threadIdx.x & 31;
    const int wave   = blockIdx.x * (blockDim.x >> 5) + (threadIdx.x >> 5);
    const int tilesN = (N + 16 * NT - 1) / (16 * NT);
    const int tilesM = M >> 4;
    if (wave >= tilesM * tilesN) return;           // uniform per-wave branch

    const int tm   = (wave / tilesN) * 16;
    const int tn0  = (wave % tilesN) * (16 * NT);
    const int half = lane >> 4;
    const int l16  = lane & 15;

    const _Float16* Ap = A + (size_t)(tm + l16) * lda + half * 8;
    const _Float16* Wp[NT];
#pragma unroll
    for (int t = 0; t < NT; ++t) {
        int n = tn0 + 16 * t + l16;
        if (n >= N) n = N - 1;                     // clamp (edge tiles, N=88)
        Wp[t] = W + (size_t)n * ldw + half * 8;
    }

    v8f acc[NT];
#pragma unroll
    for (int t = 0; t < NT; ++t) acc[t] = (v8f){0.f,0.f,0.f,0.f,0.f,0.f,0.f,0.f};

    // ping-pong fragment sets
    v16h af0 = load_frag_h(Ap);
    v16h bf0[NT];
#pragma unroll
    for (int t = 0; t < NT; ++t) bf0[t] = load_frag_h(Wp[t]);

    int k0 = 0;
    while (true) {
        const bool more1 = (k0 + 32 < K);          // uniform
        v16h af1;
        v16h bf1[NT];
        if (more1) {                               // load set 1 for k0+32
            __builtin_prefetch(Ap + k0 + 128, 0, 3);
            af1 = load_frag_h(Ap + k0 + 32);
#pragma unroll
            for (int t = 0; t < NT; ++t) bf1[t] = load_frag_h(Wp[t] + k0 + 32);
        }
#pragma unroll
        for (int t = 0; t < NT; ++t)               // compute with set 0
            acc[t] = __builtin_amdgcn_wmma_f32_16x16x32_f16(
                         false, af0, false, bf0[t], (short)0, acc[t], false, false);
        if (!more1) break;

        const bool more2 = (k0 + 64 < K);          // uniform
        if (more2) {                               // reload set 0 for k0+64
            af0 = load_frag_h(Ap + k0 + 64);
#pragma unroll
            for (int t = 0; t < NT; ++t) bf0[t] = load_frag_h(Wp[t] + k0 + 64);
        }
#pragma unroll
        for (int t = 0; t < NT; ++t)               // compute with set 1
            acc[t] = __builtin_amdgcn_wmma_f32_16x16x32_f16(
                         false, af1, false, bf1[t], (short)0, acc[t], false, false);
        if (!more2) break;
        k0 += 64;
    }

    // C/D layout: VGPR r -> row tm + r + 8*half, col tn + l16
#pragma unroll
    for (int t = 0; t < NT; ++t) {
        const int cn = tn0 + 16 * t + l16;
        if (cn < N) {
            float bv = 0.f;
            if (EPI != EPI_NONE) bv = bias[cn];
#pragma unroll
            for (int r = 0; r < 8; ++r) {
                const int cm = tm + r + 8 * half;
                float v = acc[t][r] + bv;
                if (EPI == EPI_BIAS_SIGMOID)  v = 1.f / (1.f + __expf(-v));
                if (EPI == EPI_BIAS_SOFTPLUS) v = (v > 20.f) ? v : log1pf(__expf(v));
                if (OMODE & OUT_F32) C32[(size_t)cm * ldc + cn] = v;
                if (OMODE & OUT_F16) C16[(size_t)cm * ldc + cn] = (_Float16)v;
            }
        }
    }
}

template<int EPI, int OMODE>
static void launch_gemm(const _Float16* A, int lda, const _Float16* W, int ldw,
                        const float* bias, float* C32, _Float16* C16, int ldc,
                        int M, int N, int K, hipStream_t stream)
{
    constexpr int NT = 4;
    const int tilesM = M / 16;
    const int tilesN = (N + 16 * NT - 1) / (16 * NT);
    const long threads = (long)tilesM * tilesN * 32;
    const int blocks = (int)((threads + 255) / 256);
    gemm_wmma_f16<EPI, OMODE, NT><<<blocks, 256, 0, stream>>>(
        A, lda, W, ldw, bias, C32, C16, ldc, M, N, K);
}

// f32 -> f16 bulk convert, 4 elements/thread (n % 4 == 0 at all call sites)
__global__ __launch_bounds__(256)
void cvt4_kernel(const float* __restrict__ in, _Float16* __restrict__ out, int n4)
{
    const int i = blockIdx.x * blockDim.x + threadIdx.x;
    if (i >= n4) return;
    const float4 v = ((const float4*)in)[i];
    v4h h;
    h[0] = (_Float16)v.x; h[1] = (_Float16)v.y;
    h[2] = (_Float16)v.z; h[3] = (_Float16)v.w;
    ((v4h*)out)[i] = h;
}
static void launch_cvt(const float* in, _Float16* out, int n, hipStream_t s) {
    const int n4 = n / 4;
    cvt4_kernel<<<(n4 + 255) / 256, 256, 0, s>>>(in, out, n4);
}

// ---------------------------------------------------------------------------
// Conv front-end: 3x3 SAME conv + BN(g*rsqrt(1+eps), b) + ReLU.  blockIdx.y=co
// keeps all weight indices wave-uniform (scalar loads).
// ---------------------------------------------------------------------------
__global__ __launch_bounds__(256)
void conv3x3_bn_relu(const float* __restrict__ in, const float* __restrict__ wt,
                     const float* __restrict__ cb, const float* __restrict__ gn,
                     const float* __restrict__ bt, float* __restrict__ out,
                     int Cin, int H, int Wd)
{
    const int hw = blockIdx.x * blockDim.x + threadIdx.x;
    if (hw >= H * Wd) return;
    const int co = blockIdx.y, b = blockIdx.z, Cout = gridDim.y;
    const int h = hw / Wd;
    const int x = hw - h * Wd;
    float acc = cb[co];
    for (int ci = 0; ci < Cin; ++ci) {
        const float* ip = in + (((size_t)b * Cin + ci) * H) * Wd;
        const float* wp = wt + ((size_t)co * Cin + ci) * 9;
#pragma unroll
        for (int dh = -1; dh <= 1; ++dh) {
            const int hh = h + dh;
            if (hh < 0 || hh >= H) continue;
            const float* row = ip + (size_t)hh * Wd;
#pragma unroll
            for (int dw = -1; dw <= 1; ++dw) {
                const int xx = x + dw;
                if (xx < 0 || xx >= Wd) continue;
                acc += wp[(dh + 1) * 3 + (dw + 1)] * row[xx];
            }
        }
    }
    const float v = acc * (gn[co] * rsqrtf(1.0f + 1e-5f)) + bt[co];
    out[(((size_t)b * Cout + co) * H + h) * Wd + x] = fmaxf(v, 0.0f);
}

__global__ __launch_bounds__(256)
void maxpool_w2_kernel(const float* __restrict__ in, float* __restrict__ out,
                       int Wout, int Win, int n)
{
    const int idx = blockIdx.x * blockDim.x + threadIdx.x;
    if (idx >= n) return;
    const int wo  = idx % Wout;
    const int row = idx / Wout;
    const float* p = in + (size_t)row * Win + 2 * wo;
    out[idx] = fmaxf(p[0], p[1]);
}

// (4,64,1024,57) NCHW -> (4096, 3648) token-major, directly in f16 (FC input)
__global__ __launch_bounds__(256)
void reshape_h_kernel(const float* __restrict__ in, _Float16* __restrict__ out)
{
    const int idx = blockIdx.x * blockDim.x + threadIdx.x;
    if (idx >= 4096 * 3648) return;
    const int k = idx % 3648;
    const int m = idx / 3648;
    const int b = m >> 10, t = m & 1023;
    const int c = k / 57,  f = k - c * 57;
    out[idx] = (_Float16)in[(((size_t)b * 64 + c) * 1024 + t) * 57 + f];
}

// causal depthwise conv1d (D_CONV=4) + SiLU; writes f32 (scan) + f16 (x_proj)
__global__ __launch_bounds__(256)
void dwconv_silu_kernel(const float* __restrict__ xz, const float* __restrict__ wt,
                        const float* __restrict__ cb, float* __restrict__ xc32,
                        _Float16* __restrict__ xc16)
{
    const int idx = blockIdx.x * blockDim.x + threadIdx.x;   // (b*1024+t)*1024+d
    if (idx >= 4 * 1024 * 1024) return;
    const int d = idx & 1023;
    const int t = (idx >> 10) & 1023;
    const int b = idx >> 20;
    const float* col = xz + (size_t)(b * 1024) * 2048 + d;
    const float* wd  = wt + d * 4;
    float acc = cb[d];
#pragma unroll
    for (int j = 0; j < 4; ++j) {
        const int tt = t - 3 + j;
        if (tt >= 0) acc += wd[j] * col[(size_t)tt * 2048];
    }
    const float v = acc * (1.f / (1.f + __expf(-acc)));
    xc32[idx] = v;
    xc16[idx] = (_Float16)v;
}

// ---------------------------------------------------------------------------
// Selective scan, fused: one thread per (b,d), h[16] in VGPRs.
//   h = exp(dt*A)*h + (dt*u)*B ; y = <h,C> + u*D ; ys = y * SiLU(z)   (f16 out)
// xdbl rows: [0:32)=dt_raw  [32:48)=B  [48:64)=C
// ---------------------------------------------------------------------------
__global__ __launch_bounds__(256)
void scan_kernel(const float* __restrict__ xc, const float* __restrict__ xz,
                 const float* __restrict__ xdbl, const float* __restrict__ dtv,
                 const float* __restrict__ a_log, const float* __restrict__ dpar,
                 _Float16* __restrict__ ys)
{
    const int idx = blockIdx.x * blockDim.x + threadIdx.x;
    if (idx >= 4096) return;
    const int d = idx & 1023;
    const int b = idx >> 10;
    float Am[16], h[16];
#pragma unroll
    for (int s = 0; s < 16; ++s) { Am[s] = -__expf(a_log[d * 16 + s]); h[s] = 0.f; }
    const float Dp = dpar[d];
    for (int t = 0; t < 1024; ++t) {
        const size_t row = (size_t)b * 1024 + t;
        const float dt = dtv[row * 1024 + d];
        const float u  = xc [row * 1024 + d];
        const float zv = xz [row * 2048 + 1024 + d];
        const float* bc = xdbl + row * 64 + 32;
        const float du = dt * u;
        float y = 0.f;
#pragma unroll
        for (int s = 0; s < 16; ++s) {
            h[s] = __expf(dt * Am[s]) * h[s] + du * bc[s];
            y += h[s] * bc[16 + s];
        }
        y += u * Dp;
        const float sz = zv * (1.f / (1.f + __expf(-zv)));
        ys[row * 1024 + d] = (_Float16)(y * sz);
    }
}

// ---------------------------------------------------------------------------
// Orchestration
// ---------------------------------------------------------------------------
extern "C" void kernel_launch(void* const* d_in, const int* in_sizes, int n_in,
                              void* d_out, int out_size, void* d_ws, size_t ws_size,
                              hipStream_t stream)
{
    (void)in_sizes; (void)n_in; (void)out_size; (void)ws_size;
    const float* mel       = (const float*)d_in[0];
    const float* conv1_w   = (const float*)d_in[1];
    const float* conv1_b   = (const float*)d_in[2];
    const float* bn1_g     = (const float*)d_in[3];
    const float* bn1_b     = (const float*)d_in[4];
    const float* conv2_w   = (const float*)d_in[5];
    const float* conv2_b   = (const float*)d_in[6];
    const float* bn2_g     = (const float*)d_in[7];
    const float* bn2_b     = (const float*)d_in[8];
    const float* conv3_w   = (const float*)d_in[9];
    const float* conv3_b   = (const float*)d_in[10];
    const float* bn3_g     = (const float*)d_in[11];
    const float* bn3_b     = (const float*)d_in[12];
    const float* fc_w      = (const float*)d_in[13];
    const float* fc_b      = (const float*)d_in[14];
    const float* in_proj_w = (const float*)d_in[15];
    const float* conv1d_w  = (const float*)d_in[16];
    const float* conv1d_b  = (const float*)d_in[17];
    const float* x_proj_w  = (const float*)d_in[18];
    const float* dt_proj_w = (const float*)d_in[19];
    const float* dt_proj_b = (const float*)d_in[20];
    const float* A_log     = (const float*)d_in[21];
    const float* D_param   = (const float*)d_in[22];
    const float* out_proj_w= (const float*)d_in[23];
    const float* onset_w   = (const float*)d_in[24];
    const float* onset_b   = (const float*)d_in[25];
    const float* offset_w  = (const float*)d_in[26];
    const float* offset_b  = (const float*)d_in[27];
    const float* frame_w   = (const float*)d_in[28];
    const float* frame_b   = (const float*)d_in[29];
    const float* vel_w     = (const float*)d_in[30];
    const float* vel_b     = (const float*)d_in[31];

    // ---- workspace -------------------------------------------------------
    // bufA / bufB: 30,015,488 f32 each (conv ping-pong).  After the conv
    // stage, bufA hosts the Mamba f32 tensors + the f16 staging region and
    // bufB hosts the f16 FC input.  Total ws use: ~229 MB.
    float* bufA = (float*)d_ws;
    float* bufB = bufA + 30015488ull;

    // --- conv front-end ---------------------------------------------------
    conv3x3_bn_relu<<<dim3(916, 32, 4), 256, 0, stream>>>(
        mel, conv1_w, conv1_b, bn1_g, bn1_b, bufA, 1, 1024, 229);
    conv3x3_bn_relu<<<dim3(916, 32, 4), 256, 0, stream>>>(
        bufA, conv2_w, conv2_b, bn2_g, bn2_b, bufB, 32, 1024, 229);
    maxpool_w2_kernel<<<58368, 256, 0, stream>>>(bufB, bufA, 114, 229, 14942208);
    conv3x3_bn_relu<<<dim3(456, 64, 4), 256, 0, stream>>>(
        bufA, conv3_w, conv3_b, bn3_g, bn3_b, bufB, 32, 1024, 114);
    maxpool_w2_kernel<<<58368, 256, 0, stream>>>(bufB, bufA, 57, 114, 14942208);

    _Float16* ah_fcin = (_Float16*)bufB;                 // (4096,3648) f16
    reshape_h_kernel<<<58368, 256, 0, stream>>>(bufA, ah_fcin);

    // --- Mamba f32 tensors in bufA (conv data dead from here on) ----------
    float* xz     = bufA;                 // (4096,2048) [xi | z]
    float* xc32   = bufA +  8388608;      // (4096,1024)
    float* xdbl32 = bufA + 12582912;      // (4096,64)   [dt | B | C]
    float* dtv    = bufA + 12845056;      // (4096,1024)
    // --- f16 staging region (activations + converted weights) -------------
    _Float16* hp     = (_Float16*)(bufA + 17039360);
    _Float16* ah_x    = hp;               // (4096,512)
    _Float16* ah_xc   = hp +  2097152;    // (4096,1024)
    _Float16* ah_xdbl = hp +  6291456;    // (4096,64)
    _Float16* ah_ys   = hp +  6553600;    // (4096,1024)
    _Float16* wh_fc   = hp + 10747904;    // (512,3648)
    _Float16* wh_in   = hp + 12615680;    // 4x(2048,512)
    _Float16* wh_xp   = hp + 16809984;    // 4x(64,1024)
    _Float16* wh_dt   = hp + 17072128;    // 4x(1024,32)
    _Float16* wh_op   = hp + 17203200;    // 4x(512,1024)
    _Float16* wh_hd   = hp + 19300352;    // 4x(88,512)

    // Convert weights once (after conv stage so the region is free)
    launch_cvt(fc_w,       wh_fc, 512 * 3648,        stream);
    launch_cvt(in_proj_w,  wh_in, 4 * 2048 * 512,    stream);
    launch_cvt(x_proj_w,   wh_xp, 4 * 64 * 1024,     stream);
    launch_cvt(dt_proj_w,  wh_dt, 4 * 1024 * 32,     stream);
    launch_cvt(out_proj_w, wh_op, 4 * 512 * 1024,    stream);
    launch_cvt(onset_w,  wh_hd + 0 * 45056, 88 * 512, stream);
    launch_cvt(offset_w, wh_hd + 1 * 45056, 88 * 512, stream);
    launch_cvt(frame_w,  wh_hd + 2 * 45056, 88 * 512, stream);
    launch_cvt(vel_w,    wh_hd + 3 * 45056, 88 * 512, stream);

    // FC: x(f16) = reshaped @ fc_w^T + fc_b
    launch_gemm<EPI_BIAS, OUT_F16>(ah_fcin, 3648, wh_fc, 3648, fc_b,
                                   nullptr, ah_x, 512, 4096, 512, 3648, stream);

    // --- 4 Mamba blocks ---------------------------------------------------
    for (int i = 0; i < 4; ++i) {
        const _Float16* ipw = wh_in + (size_t)i * 2048 * 512;
        const _Float16* xpw = wh_xp + (size_t)i * 64 * 1024;
        const _Float16* dtw = wh_dt + (size_t)i * 1024 * 32;
        const _Float16* opw = wh_op + (size_t)i * 512 * 1024;
        const float* c1w  = conv1d_w  + (size_t)i * 1024 * 4;
        const float* c1b  = conv1d_b  + (size_t)i * 1024;
        const float* dtb  = dt_proj_b + (size_t)i * 1024;
        const float* alog = A_log     + (size_t)i * 1024 * 16;
        const float* dpar = D_param   + (size_t)i * 1024;

        launch_gemm<EPI_NONE, OUT_F32>(ah_x, 512, ipw, 512, nullptr,
                                       xz, nullptr, 2048, 4096, 2048, 512, stream);
        dwconv_silu_kernel<<<16384, 256, 0, stream>>>(xz, c1w, c1b, xc32, ah_xc);
        launch_gemm<EPI_NONE, OUT_BOTH>(ah_xc, 1024, xpw, 1024, nullptr,
                                        xdbl32, ah_xdbl, 64, 4096, 64, 1024, stream);
        launch_gemm<EPI_BIAS_SOFTPLUS, OUT_F32>(ah_xdbl, 64, dtw, 32, dtb,
                                                dtv, nullptr, 1024, 4096, 1024, 32, stream);
        scan_kernel<<<16, 256, 0, stream>>>(xc32, xz, xdbl32, dtv, alog, dpar, ah_ys);
        launch_gemm<EPI_NONE, OUT_F16>(ah_ys, 1024, opw, 1024, nullptr,
                                       nullptr, ah_x, 512, 4096, 512, 1024, stream);
    }

    // --- heads: d_out = [onset | offset | frame | velocity], each (4,1024,88)
    float* out = (float*)d_out;
    launch_gemm<EPI_BIAS_SIGMOID, OUT_F32>(ah_x, 512, wh_hd + 0 * 45056, 512, onset_b,
                                           out,           nullptr, 88, 4096, 88, 512, stream);
    launch_gemm<EPI_BIAS_SIGMOID, OUT_F32>(ah_x, 512, wh_hd + 1 * 45056, 512, offset_b,
                                           out +  360448, nullptr, 88, 4096, 88, 512, stream);
    launch_gemm<EPI_BIAS_SIGMOID, OUT_F32>(ah_x, 512, wh_hd + 2 * 45056, 512, frame_b,
                                           out +  720896, nullptr, 88, 4096, 88, 512, stream);
    launch_gemm<EPI_BIAS, OUT_F32>        (ah_x, 512, wh_hd + 3 * 45056, 512, vel_b,
                                           out + 1081344, nullptr, 88, 4096, 88, 512, stream);
}
